// WindowAttention3d3_78752520339654
// MI455X (gfx1250) — matrix operand order repaired and checked by reading, not compiled, not run on hardware
//
#include <hip/hip_runtime.h>
#include <hip/hip_bf16.h>

// ---------------------------------------------------------------------------
// CDNA5 (gfx1250) wave32 WMMA implementation of WindowAttention3d3.
// All matmuls use v_wmma_f32_16x16x32_bf16 (f32 accumulate).
// ---------------------------------------------------------------------------

typedef __attribute__((ext_vector_type(16))) __bf16 v16bf;
typedef __attribute__((ext_vector_type(8)))  float  v8f;

#define HW      112
#define C_DIM   512
#define NH      8
#define HD      64
#define WS2     49
#define NTOK    25088      // 2*112*112 == 512 windows * 49
#define LROWS   441        // 9 * 49 K/V rows per window-head
#define LPAD    448        // padded to multiple of 32
#define QKV_LD  1536
#define CLIP_LD 1024

static __device__ inline v8f v8f_zero() {
  v8f z = {0.f, 0.f, 0.f, 0.f, 0.f, 0.f, 0.f, 0.f};
  return z;
}

static __device__ inline v8f wmma_bf16(v16bf a, v16bf b, v8f c) {
  return __builtin_amdgcn_wmma_f32_16x16x32_bf16(false, a, false, b, (short)0, c,
                                                 false, false);
}

// A fragment (16x32 bf16), source LDS row-major [rows][ld].
// lane L: M = row0 + (L&15); K = kbase + (L>>4)*16 + 2j + e.
static __device__ inline v16bf frag_a_lds(const __bf16* p, int row0, int ld, int kbase) {
  const int lane = threadIdx.x & 31;
  const __bf16* r = p + (row0 + (lane & 15)) * ld + kbase + ((lane >> 4) << 4);
  v16bf a;
#pragma unroll
  for (int j = 0; j < 8; ++j) {
    const int k = j << 1;
    a[2 * j]     = r[k];
    a[2 * j + 1] = r[k + 1];
  }
  return a;
}

// B fragment (32x16 bf16), source LDS K-major [K][ld] (row = k, col = n).
// lane L: N = n0 + (L&15); K = kbase + (L>>4)*16 + 2j + e.
static __device__ inline v16bf frag_b_lds(const __bf16* p, int kbase, int n0, int ld) {
  const int lane = threadIdx.x & 31;
  const __bf16* c = p + (kbase + ((lane >> 4) << 4)) * ld + n0 + (lane & 15);
  v16bf b;
#pragma unroll
  for (int j = 0; j < 8; ++j) {
    b[2 * j]     = c[(2 * j) * ld];
    b[2 * j + 1] = c[(2 * j + 1) * ld];
  }
  return b;
}

// B fragment where the operand is stored N-major [N][ld] (i.e. K_all rows),
// so B[k][n] = src[n][k] -- contiguous along k, no transpose needed.
static __device__ inline v16bf frag_bt_lds(const __bf16* p, int n0, int ld, int kbase) {
  const int lane = threadIdx.x & 31;
  const __bf16* r = p + (n0 + (lane & 15)) * ld + kbase + ((lane >> 4) << 4);
  v16bf b;
#pragma unroll
  for (int j = 0; j < 8; ++j) {
    b[2 * j]     = r[2 * j];
    b[2 * j + 1] = r[2 * j + 1];
  }
  return b;
}

// ---------------------------------------------------------------------------
// Generic bf16 WMMA GEMM:  C[M x Ncols] = A[M x K] * B[K x .](cols n_start..)
// Block tile 128x128, 128 threads = 4 waves (2x2), each wave a 64x64 tile:
// 16 WMMAs per 32-wide K-step from 8 fragment loads (0.5 fragment loads / WMMA).
// A_IS_F32: A is float (converted to bf16 at LDS stage) else bf16.
// OUT_F32 : write float (final output) else bf16 (intermediate).
// ---------------------------------------------------------------------------
template <bool A_IS_F32, bool OUT_F32>
__global__ __launch_bounds__(128) void gemm_bf16_wmma(
    const void* __restrict__ A_, const float* __restrict__ Bw,
    const float* __restrict__ bias, void* __restrict__ Cout, int K, int ldb,
    int n_start, int ldc) {
  __shared__ __bf16 As[128 * 32];   // 8 KB
  __shared__ __bf16 Bs[32 * 128];   // 8 KB

  const int tid  = threadIdx.x;
  const int wave = tid >> 5;
  const int wm   = wave >> 1;   // 0..1
  const int wn   = wave & 1;    // 0..1
  const long rowBase = (long)blockIdx.y * 128;
  const int  colBase = blockIdx.x * 128;

  const float*  Af = (const float*)A_;
  const __bf16* Ab = (const __bf16*)A_;

  v8f acc[4][4];
#pragma unroll
  for (int i = 0; i < 4; ++i)
#pragma unroll
    for (int j = 0; j < 4; ++j) acc[i][j] = v8f_zero();

  for (int k0 = 0; k0 < K; k0 += 32) {
    // Stage A tile (128x32) -> bf16 LDS
#pragma unroll
    for (int idx = tid; idx < 128 * 32; idx += 128) {
      const int i = idx >> 5, j = idx & 31;
      const long off = (rowBase + i) * (long)K + k0 + j;
      const float v = A_IS_F32 ? Af[off] : (float)Ab[off];
      As[idx] = (__bf16)v;
    }
    // Stage B tile (32x128) -> bf16 LDS (K-major)
#pragma unroll
    for (int idx = tid; idx < 32 * 128; idx += 128) {
      const int ki = idx >> 7, n = idx & 127;
      Bs[idx] = (__bf16)Bw[(long)(k0 + ki) * ldb + n_start + colBase + n];
    }
    __syncthreads();

    v16bf af[4], bf[4];
#pragma unroll
    for (int t = 0; t < 4; ++t) {
      af[t] = frag_a_lds(As, wm * 64 + t * 16, 32, 0);
      bf[t] = frag_b_lds(Bs, 0, wn * 64 + t * 16, 128);
    }
#pragma unroll
    for (int tm = 0; tm < 4; ++tm)
#pragma unroll
      for (int tn = 0; tn < 4; ++tn)
        acc[tm][tn] = wmma_bf16(af[tm], bf[tn], acc[tm][tn]);
    __syncthreads();
  }

  // Epilogue: bias add, write out. C/D layout: lane&15 = N, (lane>>4)*8+v = M.
  const int lane  = tid & 31;
  const int nlane = lane & 15;
  const int mbase = (lane >> 4) << 3;
#pragma unroll
  for (int tm = 0; tm < 4; ++tm) {
#pragma unroll
    for (int tn = 0; tn < 4; ++tn) {
#pragma unroll
      for (int v = 0; v < 8; ++v) {
        const long row = rowBase + wm * 64 + tm * 16 + mbase + v;
        const int  col = colBase + wn * 64 + tn * 16 + nlane;
        const float r = acc[tm][tn][v] + bias[n_start + col];
        if (OUT_F32)
          ((float*)Cout)[row * ldc + col] = r;
        else
          ((__bf16*)Cout)[row * ldc + col] = (__bf16)r;
      }
    }
  }
}

// ---------------------------------------------------------------------------
// Gather one 448x64 K or V operand (window-head) into LDS, N-major.
// col_off = 512 for K, 1024 for V (offsets into qkv buffer / clip buffers).
// ---------------------------------------------------------------------------
static __device__ void load_kv_rows(__bf16* dst, int col_off, int b, int wh, int ww,
                                    int win, int h, const __bf16* qkv,
                                    const float* pool0, const __bf16* c1,
                                    const __bf16* c2, const __bf16* c3) {
  for (int i = threadIdx.x; i < LPAD * 64; i += 256) {
    const int row = i >> 6, d = i & 63;
    float v = 0.f;
    if (row < WS2) {                       // self window
      const int t = row;
      const int y = wh * 7 + t / 7, x = ww * 7 + t % 7;
      const long tok = ((long)b * HW + y) * HW + x;
      v = (float)qkv[tok * QKV_LD + col_off + h * HD + d];
    } else if (row < 245) {                // 4 rolled windows, shifts +-3
      const int p = (row - 49) / 49, t = (row - 49) % 49;
      const int sy = (p < 2) ? -3 : 3;
      const int sx = (p & 1) ? 3 : -3;
      const int y = (wh * 7 + t / 7 - sy + HW) % HW;
      const int x = (ww * 7 + t % 7 - sx + HW) % HW;
      const long tok = ((long)b * HW + y) * HW + x;
      v = (float)qkv[tok * QKV_LD + col_off + h * HD + d];
    } else if (row < 294) {                // pool0: raw reshape (8,49,64)
      const int p = row - 245;
      v = pool0[(long)win * 25088 + h * 3136 + p * HD + d];
    } else if (row < LROWS) {              // clips 1..3 (k at 0, v at 512)
      const int ci = (row - 294) / 49, p = (row - 294) % 49;
      const __bf16* cb = (ci == 0) ? c1 : ((ci == 1) ? c2 : c3);
      v = (float)cb[((long)win * WS2 + p) * CLIP_LD + (col_off - 512) + h * HD + d];
    }
    dst[i] = (__bf16)v;
  }
}

// ---------------------------------------------------------------------------
// Attention: one block per (window, head). 256 threads = 8 waves.
// S = (Q*scale) Kall^T -> softmax(f32) -> O = P Vall.
// ---------------------------------------------------------------------------
__global__ __launch_bounds__(256) void win_attn(
    const __bf16* __restrict__ qkv, const float* __restrict__ pool0,
    const __bf16* __restrict__ ckv1, const __bf16* __restrict__ ckv2,
    const __bf16* __restrict__ ckv3, __bf16* __restrict__ out) {
  __shared__ __bf16 Qs[64 * 64];        //  8 KB
  __shared__ __bf16 KVs[LPAD * 64];     // 56 KB (K, then reused for V)
  __shared__ float  Ss[64 * LPAD];      // 112 KB
  __shared__ __bf16 Ps[64 * LPAD];      // 56 KB

  const int tid  = threadIdx.x;
  const int wave = tid >> 5;
  const int lane = tid & 31;
  const int win  = blockIdx.x >> 3;
  const int h    = blockIdx.x & 7;
  const int b    = win >> 8;
  const int wh   = (win >> 4) & 15;
  const int ww   = win & 15;

  // zero P (covers col/row padding)
  for (int i = tid; i < 64 * LPAD; i += 256) Ps[i] = (__bf16)0.f;

  // Q, pre-scaled by hd^-0.5 = 0.125
  for (int i = tid; i < 64 * 64; i += 256) {
    const int t = i >> 6, d = i & 63;
    float v = 0.f;
    if (t < WS2) {
      const int y = wh * 7 + t / 7, x = ww * 7 + t % 7;
      const long tok = ((long)b * HW + y) * HW + x;
      v = (float)qkv[tok * QKV_LD + h * HD + d] * 0.125f;
    }
    Qs[i] = (__bf16)v;
  }
  load_kv_rows(KVs, 512, b, wh, ww, win, h, qkv, pool0, ckv1, ckv2, ckv3);
  __syncthreads();

  // ---- S = Q @ K^T : 4 x 28 tiles of 16x16, 14 per wave, Kdim = 64 ----
  const int nlane = lane & 15;
  const int mbase = (lane >> 4) << 3;
#pragma unroll
  for (int i = 0; i < 14; ++i) {
    const int tt = wave * 14 + i;
    const int tm = tt & 3, tn = tt >> 2;
    v8f acc = v8f_zero();
#pragma unroll
    for (int ks = 0; ks < 2; ++ks) {
      const v16bf a  = frag_a_lds(Qs, tm * 16, 64, ks * 32);
      const v16bf bt = frag_bt_lds(KVs, tn * 16, 64, ks * 32);
      acc = wmma_bf16(a, bt, acc);
    }
#pragma unroll
    for (int v = 0; v < 8; ++v)
      Ss[(tm * 16 + mbase + v) * LPAD + tn * 16 + nlane] = acc[v];
  }
  __syncthreads();

  // ---- overwrite K with V; then f32 softmax over 441 valid columns ----
  load_kv_rows(KVs, 1024, b, wh, ww, win, h, qkv, pool0, ckv1, ckv2, ckv3);

  for (int r = wave; r < WS2; r += 8) {
    float m = -1e30f;
    for (int c = lane; c < LROWS; c += 32) m = fmaxf(m, Ss[r * LPAD + c]);
#pragma unroll
    for (int off = 16; off; off >>= 1) m = fmaxf(m, __shfl_xor(m, off, 32));
    float s = 0.f;
    for (int c = lane; c < LROWS; c += 32) {
      const float e = __expf(Ss[r * LPAD + c] - m);
      Ss[r * LPAD + c] = e;
      s += e;
    }
#pragma unroll
    for (int off = 16; off; off >>= 1) s += __shfl_xor(s, off, 32);
    const float inv = 1.f / s;
    for (int c = lane; c < LROWS; c += 32)
      Ps[r * LPAD + c] = (__bf16)(Ss[r * LPAD + c] * inv);
  }
  __syncthreads();

  // ---- O = P @ V : 4 x 4 tiles, 2 per wave, Kdim = 448 ----
#pragma unroll
  for (int i = 0; i < 2; ++i) {
    const int tt = wave * 2 + i;
    const int tm = tt >> 2, tn = tt & 3;
    v8f acc = v8f_zero();
    for (int ks = 0; ks < 14; ++ks) {
      const v16bf a = frag_a_lds(Ps, tm * 16, LPAD, ks * 32);
      const v16bf bv = frag_b_lds(KVs, ks * 32, tn * 16, 64);
      acc = wmma_bf16(a, bv, acc);
    }
#pragma unroll
    for (int v = 0; v < 8; ++v) {
      const int t = tm * 16 + mbase + v;
      if (t < WS2)
        out[((long)win * WS2 + t) * C_DIM + h * HD + tn * 16 + nlane] = (__bf16)acc[v];
    }
  }
}

// ---------------------------------------------------------------------------
extern "C" void kernel_launch(void* const* d_in, const int* in_sizes, int n_in,
                              void* d_out, int out_size, void* d_ws, size_t ws_size,
                              hipStream_t stream) {
  const float* x      = (const float*)d_in[0];
  const float* pool0  = (const float*)d_in[1];
  const float* clip1  = (const float*)d_in[2];
  const float* clip2  = (const float*)d_in[3];
  const float* clip3  = (const float*)d_in[4];
  const float* qkv_w  = (const float*)d_in[5];
  const float* qkv_b  = (const float*)d_in[6];
  const float* proj_w = (const float*)d_in[7];
  const float* proj_b = (const float*)d_in[8];

  char* ws = (char*)d_ws;
  __bf16* qkv = (__bf16*)ws;                  ws += (size_t)NTOK * QKV_LD * 2;
  __bf16* ckv[3];
  for (int i = 0; i < 3; ++i) { ckv[i] = (__bf16*)ws; ws += (size_t)NTOK * CLIP_LD * 2; }
  __bf16* attn = (__bf16*)ws;                 ws += (size_t)NTOK * C_DIM * 2;

  const dim3 blk(128);
  // 1) main QKV: (25088x512)x(512x1536) -> bf16 qkv buffer
  gemm_bf16_wmma<true, false><<<dim3(QKV_LD / 128, NTOK / 128), blk, 0, stream>>>(
      x, qkv_w, qkv_b, qkv, C_DIM, QKV_LD, 0, QKV_LD);
  // 2) clip K/V only (columns 512..1535)
  const float* clips[3] = {clip1, clip2, clip3};
  for (int i = 0; i < 3; ++i)
    gemm_bf16_wmma<true, false><<<dim3(CLIP_LD / 128, NTOK / 128), blk, 0, stream>>>(
        clips[i], qkv_w, qkv_b, ckv[i], C_DIM, QKV_LD, 512, CLIP_LD);
  // 3) windowed attention with rolled / pooled / clip K,V
  win_attn<<<512 * NH, 256, 0, stream>>>(qkv, pool0, ckv[0], ckv[1], ckv[2], attn);
  // 4) output projection -> f32 d_out
  gemm_bf16_wmma<false, true><<<dim3(C_DIM / 128, NTOK / 128), blk, 0, stream>>>(
      attn, proj_w, proj_b, d_out, C_DIM, C_DIM, 0, C_DIM);
}
